// RBFN_6760278524220
// MI455X (gfx1250) — matrix operand order repaired and checked by reading, not compiled, not run on hardware
//
#include <hip/hip_runtime.h>

// ---------------------------------------------------------------------------
// RBFN fused forward for gfx1250 (MI455X), wave32 + WMMA bf16 + async->LDS.
//   out = concat([X, exp(-beta*(|x|^2+|c|^2-2 X C^T))]) @ W^T + b
// Sizes: N=16384, K=256, M(centers)=2048, OUT=128.
// ---------------------------------------------------------------------------

typedef __attribute__((ext_vector_type(16))) __bf16       v16bf;
typedef __attribute__((ext_vector_type(8)))  float        v8f;
typedef __attribute__((ext_vector_type(4)))  float        f32x4;
typedef __attribute__((ext_vector_type(4)))  unsigned int u32x4;
typedef __attribute__((address_space(3)))    __bf16       lds_bf16;

#define WMMA_BF16(A, B, C) \
    __builtin_amdgcn_wmma_f32_16x16x32_bf16(false, (A), false, (B), (short)0, (C), false, false)

union BFrag { u32x4 u[2]; v16bf v; };   // 8 VGPRs = 16 bf16 fragment
union BF8   { __bf16 h[8]; u32x4 u; };  // 8 bf16 pack for b128 stores

static_assert(sizeof(v16bf) == 32, "v16bf size");

__device__ __forceinline__ u32x4 pack_bf8(f32x4 a, f32x4 b) {
    BF8 p;
#pragma unroll
    for (int j = 0; j < 4; ++j) { p.h[j] = (__bf16)a[j]; p.h[4 + j] = (__bf16)b[j]; }
    return p.u;
}

// Split 16 f32 into bf16 hi + bf16 residual lo fragments (fp32-accurate GEMM)
__device__ __forceinline__ void split16(const float* f, v16bf& hi, v16bf& lo) {
#pragma unroll
    for (int e = 0; e < 16; ++e) {
        __bf16 h = (__bf16)f[e];
        hi[e] = h;
        lo[e] = (__bf16)(f[e] - (float)h);
    }
}

__device__ __forceinline__ void fill4(float* f, f32x4 v) {
#pragma unroll
    for (int j = 0; j < 4; ++j) f[j] = v[j];
}

// Async global->LDS 16B copy (gfx1250; ASYNCcnt-tracked).
__device__ __forceinline__ void async_ld_b128(void* lds_dst, const void* gsrc) {
    unsigned loff = (unsigned)(unsigned long long)(lds_bf16*)lds_dst;
    asm volatile("global_load_async_to_lds_b128 %0, %1, off"
                 :: "v"(loff), "v"(gsrc) : "memory");
}
__device__ __forceinline__ void wait_async0() {
    asm volatile("s_wait_asynccnt 0" ::: "memory");
}

// ---------------------------------------------------------------------------
// Kernel 0: row sum-of-squares for batches (16384x256) and centers (2048x256)
// into ws: ws[0..16384) = |x|^2, ws[16384..18432) = |c|^2. One wave per row.
// ---------------------------------------------------------------------------
__global__ void __launch_bounds__(256) rbfn_rowsq(const float* __restrict__ batches,
                                                  const float* __restrict__ centers,
                                                  float* __restrict__ ws) {
    const int wid  = (int)((blockIdx.x * blockDim.x + threadIdx.x) >> 5);
    const int lane = (int)(threadIdx.x & 31);
    const float* src;
    float* dst;
    if (wid < 16384) {
        src = batches + (size_t)wid * 256;  dst = ws + wid;
    } else {
        const int c = wid - 16384;
        if (c >= 2048) return;
        src = centers + (size_t)c * 256;    dst = ws + 16384 + c;
    }
    float s = 0.f;
#pragma unroll
    for (int j = 0; j < 8; ++j) { float v = src[lane + 32 * j]; s += v * v; }
#pragma unroll
    for (int off = 16; off > 0; off >>= 1) s += __shfl_xor(s, off, 32);
    if (lane == 0) *dst = s;
}

// ---------------------------------------------------------------------------
// Kernel 0b: one-time f32->bf16 conversion of centers [2048x256] and the
// radial slice of W ([128][2048] = W[:,256:2304]) into workspace.
// 98304 threads, 8 elements each.
// ---------------------------------------------------------------------------
__global__ void __launch_bounds__(256) rbfn_prep(const float* __restrict__ centers,
                                                 const float* __restrict__ Wm,
                                                 unsigned short* __restrict__ cbf_raw,
                                                 unsigned short* __restrict__ wbf_raw) {
    __bf16* cbf = (__bf16*)cbf_raw;
    __bf16* wbf = (__bf16*)wbf_raw;
    const int id = (int)(blockIdx.x * blockDim.x + threadIdx.x);
    if (id < 65536) {                       // centers: 524288 f32 / 8
        const f32x4* src = (const f32x4*)centers + (size_t)id * 2;
        *(u32x4*)(cbf + (size_t)id * 8) = pack_bf8(src[0], src[1]);
    } else if (id < 98304) {                // W radial: 262144 f32 / 8
        const int j = id - 65536;
        const int o = j >> 8;               // 256 groups of 8 per out-row
        const int k = (j & 255) * 8;
        const f32x4* src = (const f32x4*)(Wm + (size_t)o * 2304 + 256 + k);
        *(u32x4*)(wbf + (size_t)o * 2048 + k) = pack_bf8(src[0], src[1]);
    }
}

// ---------------------------------------------------------------------------
// Main fused kernel. 256 threads = 8 waves; wave w owns batch rows
// [blockIdx.x*128 + 16w, +16). Grid = 128.
// ---------------------------------------------------------------------------
__global__ void __launch_bounds__(256) rbfn_fused(const float* __restrict__ batches,
                                                  const float* __restrict__ centers,
                                                  const float* __restrict__ beta,
                                                  const float* __restrict__ Wm,
                                                  const float* __restrict__ bias,
                                                  float* __restrict__ out,
                                                  const float* __restrict__ ws,
                                                  const unsigned short* __restrict__ cbf_raw,
                                                  const unsigned short* __restrict__ wbf_raw,
                                                  const int use_pre) {
    // LDS: centers tile (32 rows x 256 K, bf16, pitch 264), W radial slice
    // (128 out x 32 c, bf16, pitch 40), per-wave radial slabs (16x32 bf16).
    __shared__ __align__(16) __bf16 sC[32][264];   // 16896 B
    __shared__ __align__(16) __bf16 sW[128][40];   // 10240 B
    __shared__ __align__(16) __bf16 sR[8][16][32]; //  8192 B

    const __bf16* cbf = (const __bf16*)cbf_raw;
    const __bf16* wbf = (const __bf16*)wbf_raw;

    const int t    = (int)threadIdx.x;
    const int w    = t >> 5;        // wave 0..7
    const int lane = t & 31;
    const int g    = lane >> 4;     // half-wave 0/1
    const int m    = lane & 15;     // row/col within 16
    const int r0   = (int)blockIdx.x * 128 + w * 16;

    const float* xsq = ws;
    const float* csq = ws + 16384;

    // x_sq for this lane's accumulator rows (row = r0 + i + 8g for VGPR i)
    float xs[8];
#pragma unroll
    for (int i = 0; i < 8; ++i) xs[i] = xsq[r0 + i + 8 * g];

    v8f outacc[8];
#pragma unroll
    for (int ot = 0; ot < 8; ++ot) outacc[ot] = (v8f){};

    // -----------------------------------------------------------------------
    // Phase A: out += X @ W[:, :256]^T with bf16 hi/lo split (~fp32 accurate).
    // Also caches the bf16 hi A-fragments of this wave's rows for GEMM1.
    // -----------------------------------------------------------------------
    v16bf Ahi[8];
    {
        const float* arow = batches + (size_t)(r0 + m) * 256;
#pragma unroll
        for (int kc = 0; kc < 8; ++kc) {
            // A fragment: row m, K run [kc*32+16g, +16)
            float fa[16];
            const f32x4* p0 = (const f32x4*)(arow + kc * 32 + 16 * g);
            const f32x4* p1 = (const f32x4*)(arow + kc * 32 + 16 * g + 8);
            fill4(fa + 0, p0[0]); fill4(fa + 4, p0[1]);
            fill4(fa + 8, p1[0]); fill4(fa + 12, p1[1]);
            v16bf ahi, alo;
            split16(fa, ahi, alo);
            Ahi[kc] = ahi;
#pragma unroll
            for (int ot = 0; ot < 8; ++ot) {
                // B fragment: out col o = ot*16 + m, K run [kc*32+16g, +16)
                const f32x4* wp = (const f32x4*)(Wm + (size_t)(ot * 16 + m) * 2304 + kc * 32 + 16 * g);
                float fb[16];
                fill4(fb + 0, wp[0]); fill4(fb + 4, wp[1]);
                fill4(fb + 8, wp[2]); fill4(fb + 12, wp[3]);
                v16bf bhi, blo;
                split16(fb, bhi, blo);
                outacc[ot] = WMMA_BF16(ahi, bhi, outacc[ot]);
                outacc[ot] = WMMA_BF16(ahi, blo, outacc[ot]);
                outacc[ot] = WMMA_BF16(alo, bhi, outacc[ot]);
            }
        }
    }

    // -----------------------------------------------------------------------
    // Phase B: loop over 64 center tiles of 32. Fused cross-GEMM -> exp ->
    // radial-GEMM accumulation. Staging is async global->LDS when the bf16
    // mirrors exist, else convert-in-flight.
    // -----------------------------------------------------------------------
    for (int ct = 0; ct < 64; ++ct) {
        __syncthreads();  // previous tile compute done before overwrite
        if (use_pre) {
            // centers tile: 32 rows x 512B = 1024 x 16B chunks, 4 per thread
#pragma unroll
            for (int q = 0; q < 4; ++q) {
                const int id = q * 256 + t;
                const int c  = id >> 5;           // 32 chunks per row
                const int j  = (id & 31) * 8;     // bf16 col
                async_ld_b128(&sC[c][j], cbf + (size_t)(ct * 32 + c) * 256 + j);
            }
            // W slice: 128 rows x 64B = 512 x 16B chunks, 2 per thread
#pragma unroll
            for (int q = 0; q < 2; ++q) {
                const int id = q * 256 + t;
                const int o  = id >> 2;           // 4 chunks per row
                const int j  = (id & 3) * 8;
                async_ld_b128(&sW[o][j], wbf + (size_t)o * 2048 + ct * 32 + j);
            }
            wait_async0();
        } else {
            {   // stage centers tile with conversion
                const int c  = t >> 3;
                const int kb = (t & 7) * 32;
                const f32x4* gp = (const f32x4*)(centers + (size_t)(ct * 32 + c) * 256 + kb);
#pragma unroll
                for (int q = 0; q < 4; ++q)
                    *(u32x4*)&sC[c][kb + 8 * q] = pack_bf8(gp[2 * q], gp[2 * q + 1]);
            }
            {   // stage W radial slice with conversion
                const int o  = t >> 1;
                const int kb = (t & 1) * 16;
                const f32x4* gp = (const f32x4*)(Wm + (size_t)o * 2304 + 256 + ct * 32 + kb);
#pragma unroll
                for (int q = 0; q < 2; ++q)
                    *(u32x4*)&sW[o][kb + 8 * q] = pack_bf8(gp[2 * q], gp[2 * q + 1]);
            }
        }
        __syncthreads();

        // Prefetch next tile while we compute (global_prefetch_b8)
        if (ct + 1 < 64) {
            if (use_pre) {
                __builtin_prefetch(cbf + (size_t)((ct + 1) * 32 + (t >> 3)) * 256 + (t & 7) * 32, 0, 1);
                __builtin_prefetch(wbf + (size_t)(t >> 1) * 2048 + (ct + 1) * 32, 0, 1);
            } else {
                __builtin_prefetch(centers + (size_t)((ct + 1) * 32 + (t >> 3)) * 256 + (t & 7) * 32, 0, 1);
                __builtin_prefetch(Wm + (size_t)(t >> 1) * 2304 + 256 + (ct + 1) * 32 + (t & 1) * 16, 0, 1);
            }
        }

        // GEMM1: cross[16 rows x 16 centers] per half-tile h, then radial.
#pragma unroll
        for (int h = 0; h < 2; ++h) {
            v8f c = (v8f){};
#pragma unroll
            for (int kc = 0; kc < 8; ++kc) {
                BFrag bf;  // B frag: center col n=m, K run [kc*32+16g, +16)
                bf.u[0] = *(const u32x4*)&sC[h * 16 + m][kc * 32 + 16 * g];
                bf.u[1] = *(const u32x4*)&sC[h * 16 + m][kc * 32 + 16 * g + 8];
                c = WMMA_BF16(Ahi[kc], bf.v, c);
            }
            const int cidx = ct * 32 + h * 16 + m;
            const float cs = csq[cidx];
            const float bt = beta[cidx];
#pragma unroll
            for (int i = 0; i < 8; ++i) {
                const float sq = xs[i] + cs - 2.0f * c[i];
                const float r  = __expf(-bt * sq);
                // D-layout (row i+8g, col h*16+m) -> row-major radial slab
                sR[w][i + 8 * g][h * 16 + m] = (__bf16)r;
            }
        }

        // GEMM2 radial part: out[16 x 128] += radial[16 x 32] @ Wslice^T
        BFrag ar;  // A frag from slab: row m, run [16g,+16)
        ar.u[0] = *(const u32x4*)&sR[w][m][16 * g];
        ar.u[1] = *(const u32x4*)&sR[w][m][16 * g + 8];
#pragma unroll
        for (int ot = 0; ot < 8; ++ot) {
            BFrag bw;  // B frag: out col o = ot*16+m, K run [16g, +16)
            bw.u[0] = *(const u32x4*)&sW[ot * 16 + m][16 * g];
            bw.u[1] = *(const u32x4*)&sW[ot * 16 + m][16 * g + 8];
            outacc[ot] = WMMA_BF16(ar.v, bw.v, outacc[ot]);
        }
    }

    // Epilogue: add bias, store. outacc[ot] VGPR i -> out[r0+i+8g][ot*16+m].
#pragma unroll
    for (int ot = 0; ot < 8; ++ot) {
        const float bv = bias[ot * 16 + m];
#pragma unroll
        for (int i = 0; i < 8; ++i)
            out[(size_t)(r0 + i + 8 * g) * 128 + ot * 16 + m] = outacc[ot][i] + bv;
    }
}

// ---------------------------------------------------------------------------
extern "C" void kernel_launch(void* const* d_in, const int* in_sizes, int n_in,
                              void* d_out, int out_size, void* d_ws, size_t ws_size,
                              hipStream_t stream) {
    const float* batches = (const float*)d_in[0];  // [16384, 256]
    const float* centers = (const float*)d_in[1];  // [2048, 256]
    const float* beta    = (const float*)d_in[2];  // [1, 2048]
    const float* W       = (const float*)d_in[3];  // [128, 2304]
    const float* b       = (const float*)d_in[4];  // [128]
    float*       out     = (float*)d_out;          // [16384, 128]
    float*       ws      = (float*)d_ws;

    // ws layout: [0,16384) x_sq f32 | [16384,18432) c_sq f32 |
    //            centers bf16 [2048*256] | W radial bf16 [128*2048]
    const size_t off_cbf = 18432u * sizeof(float);          // 73728 B (16-aligned)
    const size_t off_wbf = off_cbf + 2048u * 256u * 2u;     // +1 MiB
    const size_t need    = off_wbf + 128u * 2048u * 2u;     // ~1.57 MiB total
    unsigned short* cbf = (unsigned short*)((char*)d_ws + off_cbf);
    unsigned short* wbf = (unsigned short*)((char*)d_ws + off_wbf);
    const int use_pre = (ws_size >= need) ? 1 : 0;

    rbfn_rowsq<<<2304, 256, 0, stream>>>(batches, centers, ws);
    if (use_pre)
        rbfn_prep<<<384, 256, 0, stream>>>(centers, W, cbf, wbf);
    rbfn_fused<<<128, 256, 0, stream>>>(batches, centers, beta, W, b, out, ws,
                                        cbf, wbf, use_pre);
}
